// MTGViT_35802847380255
// MI455X (gfx1250) — matrix operand order repaired and checked by reading, not compiled, not run on hardware
//
#include <hip/hip_runtime.h>
#include <hip/hip_bf16.h>
#include <math.h>

// ---------------- problem constants (ViT-Base, B=64) ----------------
#define B_    64
#define CIN   3
#define IMG   224
#define PS    16
#define NSIDE 14
#define NP    196
#define T_    197
#define E_    768
#define H_    12
#define HS_   64
#define L_    12
#define FF_   3072
#define NC_   1000
#define NCP   1024        // head N padded to multiple of 64
#define BT    (B_ * T_)   // 12608 = 197*64
#define BNP   (B_ * NP)   // 12544 = 196*64

typedef __attribute__((ext_vector_type(16))) __bf16         v16bf;
typedef __attribute__((ext_vector_type(16))) unsigned short v16u;
typedef __attribute__((ext_vector_type(8)))  float          v8f;

// f32 -> bf16 bits (round-to-nearest-even)
__device__ __forceinline__ unsigned short f2bf(float x) {
    unsigned u = __float_as_uint(x);
    unsigned r = u + 0x7FFFu + ((u >> 16) & 1u);
    return (unsigned short)(r >> 16);
}
__device__ __forceinline__ unsigned pack2(float a, float b) {
    return (unsigned)f2bf(a) | ((unsigned)f2bf(b) << 16);
}
__device__ __forceinline__ uint4 pack8(float4 a, float4 b) {
    return make_uint4(pack2(a.x, a.y), pack2(a.z, a.w),
                      pack2(b.x, b.y), pack2(b.z, b.w));
}
__device__ __forceinline__ v16bf asbf(v16u v) { return __builtin_bit_cast(v16bf, v); }

union V16 { v16u v; uint4 q[2]; unsigned u[8]; };

__device__ __forceinline__ v8f wmma_bf16(v16bf a, v16bf b, v8f c) {
    return __builtin_amdgcn_wmma_f32_16x16x32_bf16(false, a, false, b, (short)0, c,
                                                   false, false);
}

// Fragment-major layouts (contiguous k order used by this code, wave32):
//  A 16x32 tile : lane = (m&15) + 16*(k>>4),      idx = k & 15
//  B 32x16 tile : lane = (n&15) + 16*(k>>4),      idx = k & 15
// A whole fragment is one 32B (v16u) LDS read per lane.
// Global swizzled-B layout: [n-block][k-step][n-subtile(4)][lane(32)][16] bf16
//   -> one contiguous 4KB chunk per 64x32 weight tile.

// ---------------- Tensor Data Mover copy of one 4KB B tile ----------------
#if __has_builtin(__builtin_amdgcn_tensor_load_to_lds) && \
    __has_builtin(__builtin_amdgcn_s_wait_tensorcnt)
#define USE_TDM 1
typedef __attribute__((ext_vector_type(4))) unsigned tdm_v4u;
typedef __attribute__((ext_vector_type(8))) int      tdm_v8i;
typedef __attribute__((ext_vector_type(4))) int      tdm_v4i;

// D# per ISA 8.3/8.4: count=1, type=2("image"), data_size=3 (8B units),
// 1-D tensor/tile of 512 units (4KB), stride 512.
__device__ __forceinline__ void tdm_load_4k(void* lds_dst, const void* gsrc) {
    unsigned lds_addr =
        (unsigned)(unsigned long long)(__attribute__((address_space(3))) void*)lds_dst;
    unsigned long long ga = (unsigned long long)gsrc;
    tdm_v4u g0;
    g0[0] = 1u;                                              // count=1, user desc
    g0[1] = lds_addr;                                        // lds_addr
    g0[2] = (unsigned)ga;                                    // global_addr[31:0]
    g0[3] = (unsigned)((ga >> 32) & 0x01FFFFFFull) | (2u << 30);  // addr[56:32]|type=2
    tdm_v8i g1;
    g1[0] = (int)(3u << 16);          // workgroup_mask=0, data_size=3 (8B)
    g1[1] = (int)(512u << 16);        // tensor_dim0[15:0] (bits 79:64 slot)
    g1[2] = 0;                        // tensor_dim0[31:16]=0, tensor_dim1=0
    g1[3] = (int)(512u << 16);        // tile_dim0 = 512 units
    g1[4] = 0;                        // tile_dim1/2 unused
    g1[5] = 512;                      // tensor_dim0_stride = 512 units
    g1[6] = 0;
    g1[7] = 0;
    tdm_v4i z4 = {0, 0, 0, 0};
    tdm_v8i z8 = {0, 0, 0, 0, 0, 0, 0, 0};
    __builtin_amdgcn_tensor_load_to_lds(g0, g1, z4, z4, z8, 0);
}
#else
#define USE_TDM 0
#endif

// ---------------- generic fused WMMA GEMM ----------------
// C[M,N] = act(A[M,K] @ Bsw + bias) (+residual). A f32, B pre-swizzled bf16.
// Block tile 128x64x32; each wave computes a 32x32 sub-tile (4 WMMAs / k-step).
#define GBM 128
#define GBN 64
#define GBK 32

__device__ __forceinline__ void stage_a_tile(v16u* Asf, const float* A, int m0, int kb,
                                             int M, int lda) {
    int tid = threadIdx.x;
    int m = tid >> 1, k0 = (tid & 1) * 16;       // 128 rows x 2 chunks of 16 k
    float4 f0 = make_float4(0, 0, 0, 0), f1 = f0, f2 = f0, f3 = f0;
    if (m0 + m < M) {
        const float4* p = (const float4*)(A + (long)(m0 + m) * lda + kb + k0);
        f0 = p[0]; f1 = p[1]; f2 = p[2]; f3 = p[3];
    }
    // k0+0..15 -> lane half k0>>4; k0+0..7 at idx 0, k0+8..15 at idx 8
    uint4 lo = pack8(f0, f1);
    uint4 hi = pack8(f2, f3);
    int sub  = m >> 4;
    int lane = m & 15;
    *(uint4*)((unsigned short*)&Asf[(sub << 5) + lane + (k0 & 16)])     = lo;
    *(uint4*)((unsigned short*)&Asf[(sub << 5) + lane + (k0 & 16)] + 8) = hi;
}

__device__ __forceinline__ void stage_b_tile_fallback(v16u* Bsf,
                                                      const unsigned short* Btile) {
    int tid = threadIdx.x;                 // 256 threads x 16B = 4KB
    ((uint4*)Bsf)[tid] = ((const uint4*)Btile)[tid];
}

__global__ __launch_bounds__(256) void gemm_kernel(
    const float* __restrict__ A, const unsigned short* __restrict__ Bsw,
    float* __restrict__ C,
    const float* __restrict__ bias, const float* __restrict__ res,
    int M, int N, int K, int lda, int ldc,
    long sA, long sB, long sC, int gelu)
{
    __shared__ v16u Asf[2][256];   // 8 m-subtiles x 32 lanes, double buffered (16 KB)
    __shared__ v16u Bsf[2][128];   // 4 n-subtiles x 32 lanes, double buffered (8 KB)

    int z = blockIdx.z;
    A   += (long)z * sA;
    Bsw += (long)z * sB;
    C   += (long)z * sC;

    int m0 = blockIdx.y * GBM;
    int n0 = blockIdx.x * GBN;
    int tid = threadIdx.x, lane = tid & 31, wave = tid >> 5;
    int wm = wave & 3;            // rows wm*32 .. wm*32+31 (subtiles wm*2, wm*2+1)
    int wn = wave >> 2;           // cols wn*32 .. wn*32+31 (subtiles wn*2, wn*2+1)
    int ksteps = K >> 5;
    const unsigned short* Bblk = Bsw + (((long)(n0 >> 6) * ksteps) << 11); // *2048

    v8f acc[2][2] = { { {0,0,0,0,0,0,0,0}, {0,0,0,0,0,0,0,0} },
                      { {0,0,0,0,0,0,0,0}, {0,0,0,0,0,0,0,0} } };

    stage_a_tile(Asf[0], A, m0, 0, M, lda);
#if USE_TDM
    if (wave == 0) tdm_load_4k(&Bsf[0][0], Bblk);
    if (wave == 0) __builtin_amdgcn_s_wait_tensorcnt(0);
#else
    stage_b_tile_fallback(Bsf[0], Bblk);
#endif
    __syncthreads();

    int buf = 0;
    for (int kb = 0; kb < K; kb += GBK) {
        int nxt = kb + GBK;
        if (nxt < K) {
            stage_a_tile(Asf[buf ^ 1], A, m0, nxt, M, lda);
#if USE_TDM
            if (wave == 0) tdm_load_4k(&Bsf[buf ^ 1][0], Bblk + ((long)(nxt >> 5) << 11));
#else
            stage_b_tile_fallback(Bsf[buf ^ 1], Bblk + ((long)(nxt >> 5) << 11));
#endif
        }
        v16bf b0 = asbf(Bsf[buf][((wn * 2 + 0) << 5) + lane]);
        v16bf b1 = asbf(Bsf[buf][((wn * 2 + 1) << 5) + lane]);
        v16bf a0 = asbf(Asf[buf][((wm * 2 + 0) << 5) + lane]);
        v16bf a1 = asbf(Asf[buf][((wm * 2 + 1) << 5) + lane]);
        acc[0][0] = wmma_bf16(a0, b0, acc[0][0]);
        acc[0][1] = wmma_bf16(a0, b1, acc[0][1]);
        acc[1][0] = wmma_bf16(a1, b0, acc[1][0]);
        acc[1][1] = wmma_bf16(a1, b1, acc[1][1]);
#if USE_TDM
        if (wave == 0) __builtin_amdgcn_s_wait_tensorcnt(0);
#endif
        __syncthreads();
        buf ^= 1;
    }

    int mh = (lane >> 4) << 3;
#pragma unroll
    for (int u = 0; u < 2; ++u) {
        int n = n0 + wn * 32 + u * 16 + (lane & 15);
        if (n < N) {
            float bv = bias ? bias[n] : 0.f;
#pragma unroll
            for (int t = 0; t < 2; ++t) {
#pragma unroll
                for (int v = 0; v < 8; ++v) {
                    int m = m0 + wm * 32 + t * 16 + mh + v;
                    if (m < M) {
                        float val = acc[t][u][v] + bv;
                        if (gelu) val = 0.5f * val * (1.f + erff(val * 0.70710678f));
                        if (res)  val += res[(long)m * ldc + n];
                        C[(long)m * ldc + n] = val;
                    }
                }
            }
        }
    }
}

// ---------------- weight pre-swizzle: f32 [KxN] -> bf16 fragment-major ----------------
// transB: element (k,n) at W[n*ld + k].  n >= N zero-padded (Npad multiple of 64).
__global__ __launch_bounds__(256) void swizzle_b_kernel(
    const float* __restrict__ W, unsigned short* __restrict__ S,
    int K, int N, int Npad, int ld, int transB, long sW, long sS)
{
    const float* Wz = W + (long)blockIdx.y * sW;
    unsigned short* Sz = S + (long)blockIdx.y * sS;
    long jobs = (long)Npad * (K >> 3);
    long i = (long)blockIdx.x * 256 + threadIdx.x;
    if (i >= jobs) return;
    int n  = (int)(i % Npad);
    int k0 = (int)(i / Npad) * 8;
    float vals[8];
    if (n < N) {
        if (transB) {
            const float4* p = (const float4*)(Wz + (long)n * ld + k0);
            float4 fa = p[0], fb = p[1];
            vals[0] = fa.x; vals[1] = fa.y; vals[2] = fa.z; vals[3] = fa.w;
            vals[4] = fb.x; vals[5] = fb.y; vals[6] = fb.z; vals[7] = fb.w;
        } else {
#pragma unroll
            for (int j = 0; j < 8; ++j) vals[j] = Wz[(long)(k0 + j) * ld + n];
        }
    } else {
#pragma unroll
        for (int j = 0; j < 8; ++j) vals[j] = 0.f;
    }
    uint4 pk = make_uint4(pack2(vals[0], vals[1]), pack2(vals[2], vals[3]),
                          pack2(vals[4], vals[5]), pack2(vals[6], vals[7]));
    int nb = n >> 6, kt = k0 >> 5, sub = (n & 63) >> 4;
    int lane = (n & 15) + ((((k0 & 31) >> 4) & 1) << 4);
    int idx0 = k0 & 15;
    long off = ((((long)nb * (K >> 5) + kt) << 2) + sub) * 512 + lane * 16 + idx0;
    *(uint4*)(Sz + off) = pk;
}

// ---------------- fused attention: one workgroup per (b,h) ----------------
#define TP 208
#define VP 224
#define RC 64
#define SS 209

__device__ __forceinline__ v16bf load_bt(const unsigned short* kb_, int srow, int ks) {
    int lane = threadIdx.x & 31;
    const v16u* p = (const v16u*)(kb_ + (srow + (lane & 15)) * HS_ + ks * 32 +
                                  ((lane >> 4) << 4));
    return asbf(*p);
}

__global__ __launch_bounds__(256) void attention_kernel(
    const float* __restrict__ Q, const float* __restrict__ Kp,
    const float* __restrict__ V, float* __restrict__ O)
{
    __shared__ __align__(32) unsigned short kbuf[TP * HS_];  // 26 KB row-major [t][d]
    __shared__ v16u  Vf[7 * 4 * 32];                         // 28 KB frag-major
    __shared__ v16u  Qf[4 * 2 * 32];                         //  8 KB frag-major
    __shared__ float S[RC * SS];                             // 52.3 KB
    __shared__ v16u  Pf[4 * 7 * 32];                         // 28 KB frag-major

    int bh = blockIdx.x;
    int h = bh % H_, b = bh / H_;
    long base = (long)h * BT * HS_ + (long)b * T_ * HS_;
    const float* q = Q  + base;
    const float* k = Kp + base;
    const float* v = V  + base;
    int tid = threadIdx.x, lane = tid & 31, wave = tid >> 5;

    for (int i = tid; i < (TP * HS_) / 8; i += 256) {
        int t = i >> 3, d0 = (i & 7) * 8;
        float4 fa = make_float4(0, 0, 0, 0), fb = fa;
        if (t < T_) {
            const float4* p = (const float4*)(k + (long)t * HS_ + d0);
            fa = p[0]; fb = p[1];
        }
        *(uint4*)(kbuf + t * HS_ + d0) = pack8(fa, fb);
    }
    for (int i = tid; i < 64 * 14; i += 256) {
        int d = i & 63, r = i >> 6;
        int kc = r >> 1, khalf = r & 1;
        int kb0 = kc * 32 + khalf * 16;
        V16 w;
#pragma unroll
        for (int j = 0; j < 8; ++j) {
            int t0 = kb0 + 2 * j, t1 = t0 + 1;
            float e0 = (t0 < T_) ? v[(long)t0 * HS_ + d] : 0.f;
            float e1 = (t1 < T_) ? v[(long)t1 * HS_ + d] : 0.f;
            w.u[j] = pack2(e0, e1);
        }
        Vf[(kc * 4 + (d >> 4)) * 32 + (d & 15) + 16 * khalf] = w.v;
    }

    for (int chunk = 0; chunk < 4; ++chunk) {
        int row0 = chunk * RC;
        __syncthreads();

        for (int i = tid; i < 512; i += 256) {
            int m = i >> 3, k0 = (i & 7) * 8;
            int t = row0 + m;
            float4 fa = make_float4(0, 0, 0, 0), fb = fa;
            if (t < T_) {
                const float4* p = (const float4*)(q + (long)t * HS_ + k0);
                fa = p[0]; fb = p[1];
            }
            fa.x *= 0.125f; fa.y *= 0.125f; fa.z *= 0.125f; fa.w *= 0.125f;
            fb.x *= 0.125f; fb.y *= 0.125f; fb.z *= 0.125f; fb.w *= 0.125f;
            int kt = k0 >> 5, kk = k0 & 31;
            int ln = (m & 15) + (((kk >> 4) & 1) << 4);
            int idx0 = kk & 15;
            *(uint4*)((unsigned short*)&Qf[(((m >> 4) * 2 + kt) << 5) + ln] + idx0) =
                pack8(fa, fb);
        }
        __syncthreads();

        for (int tile = wave; tile < 4 * 13; tile += 8) {
            int tm = tile / 13, tn = tile % 13;
            v8f acc = {0,0,0,0,0,0,0,0};
#pragma unroll
            for (int ks = 0; ks < 2; ++ks) {
                v16bf a  = asbf(Qf[((tm * 2 + ks) << 5) + lane]);
                v16bf bt = load_bt(kbuf, tn * 16, ks);
                acc = wmma_bf16(a, bt, acc);
            }
            int n = lane & 15, mh = (lane >> 4) << 3;
            float* srow = S + (tm * 16 + mh) * SS + tn * 16 + n;
#pragma unroll
            for (int vv = 0; vv < 8; ++vv) srow[vv * SS] = acc[vv];
        }
        __syncthreads();

        if (tid < RC) {
            int gr = row0 + tid;
            float* srow = S + tid * SS;
            bool valid = gr < T_;
            float rs = 0.f;
            if (valid) {
                float s = 0.f;
                for (int c = 0; c < T_; ++c) s += srow[c];
                float mean = s / (float)T_;
                float var = 0.f, mx = -1e30f;
                for (int c = 0; c < T_; ++c) {
                    float d = srow[c] - mean;
                    var += d * d;
                    mx = fmaxf(mx, d);
                }
                float inv = rsqrtf(var / (float)T_ + 1e-5f);
                float es = 0.f;
                for (int c = 0; c < T_; ++c) {
                    float e = __expf((srow[c] - mean - mx) * inv);
                    srow[c] = e;
                    es += e;
                }
                rs = 1.f / es;
            }
            int tm = tid >> 4;
#pragma unroll
            for (int kc = 0; kc < VP / 8; ++kc) {
                int k0 = kc * 8;
                float pv[8];
#pragma unroll
                for (int j = 0; j < 8; ++j) {
                    int c = k0 + j;
                    pv[j] = (valid && c < T_) ? srow[c] * rs : 0.f;
                }
                uint4 pk = make_uint4(pack2(pv[0], pv[1]), pack2(pv[2], pv[3]),
                                      pack2(pv[4], pv[5]), pack2(pv[6], pv[7]));
                int kt = k0 >> 5, kk = k0 & 31;
                int ln = (tid & 15) + (((kk >> 4) & 1) << 4);
                int idx0 = kk & 15;
                *(uint4*)((unsigned short*)&Pf[((tm * 7 + kt) << 5) + ln] + idx0) = pk;
            }
        }
        __syncthreads();

        for (int tile = wave; tile < 16; tile += 8) {
            int tm = tile >> 2, tn = tile & 3;
            v8f acc = {0,0,0,0,0,0,0,0};
#pragma unroll
            for (int ks = 0; ks < 7; ++ks) {
                v16bf a  = asbf(Pf[((tm * 7 + ks) << 5) + lane]);
                v16bf bf = asbf(Vf[((ks * 4 + tn) << 5) + lane]);
                acc = wmma_bf16(a, bf, acc);
            }
            int n = lane & 15, mh = (lane >> 4) << 3;
#pragma unroll
            for (int vv = 0; vv < 8; ++vv) {
                int gr = row0 + tm * 16 + mh + vv;
                if (gr < T_)
                    O[((long)b * T_ + gr) * E_ + h * HS_ + tn * 16 + n] = acc[vv];
            }
        }
    }
}

// ---------------- small helpers ----------------
__global__ __launch_bounds__(256) void layernorm_kernel(
    const float* __restrict__ X, const float* __restrict__ g,
    const float* __restrict__ b, float* __restrict__ Y, int rows)
{
    int wave = threadIdx.x >> 5, lane = threadIdx.x & 31;
    int row = blockIdx.x * 8 + wave;
    if (row >= rows) return;
    const float4* x4 = (const float4*)(X + (long)row * E_);
    float s = 0.f;
    for (int c = lane; c < E_ / 4; c += 32) {
        float4 t = x4[c];
        s += t.x + t.y + t.z + t.w;
    }
    for (int o = 16; o; o >>= 1) s += __shfl_xor(s, o, 32);
    float mean = s / (float)E_;
    float var = 0.f;
    for (int c = lane; c < E_ / 4; c += 32) {
        float4 t = x4[c];
        float dx = t.x - mean, dy = t.y - mean, dz = t.z - mean, dw = t.w - mean;
        var += dx * dx + dy * dy + dz * dz + dw * dw;
    }
    for (int o = 16; o; o >>= 1) var += __shfl_xor(var, o, 32);
    float inv = rsqrtf(var / (float)E_ + 1e-5f);
    const float4* g4 = (const float4*)g;
    const float4* b4 = (const float4*)b;
    float4* y4 = (float4*)(Y + (long)row * E_);
    for (int c = lane; c < E_ / 4; c += 32) {
        float4 t = x4[c], gg = g4[c], bb = b4[c];
        y4[c] = make_float4((t.x - mean) * inv * gg.x + bb.x,
                            (t.y - mean) * inv * gg.y + bb.y,
                            (t.z - mean) * inv * gg.z + bb.z,
                            (t.w - mean) * inv * gg.w + bb.w);
    }
}

__global__ void patchify_kernel(const float* __restrict__ x, float* __restrict__ Xp) {
    long i = (long)blockIdx.x * 256 + threadIdx.x;
    if (i >= (long)BNP * E_) return;
    int  f  = (int)(i % E_);
    long bt = i / E_;
    int  t  = (int)(bt % NP);
    int  b  = (int)(bt / NP);
    int c = f >> 8, ph = (f >> 4) & 15, pw = f & 15;
    int py = t / NSIDE, px = t % NSIDE;
    Xp[i] = x[(((long)b * CIN + c) * IMG + py * PS + ph) * IMG + px * PS + pw];
}

__global__ void build_h0_kernel(const float* __restrict__ tok, const float* __restrict__ cls,
                                const float* __restrict__ pos, float* __restrict__ Hh) {
    long i = (long)blockIdx.x * 256 + threadIdx.x;
    if (i >= (long)BT * E_) return;
    int  f  = (int)(i % E_);
    long bt = i / E_;
    int  t  = (int)(bt % T_);
    int  b  = (int)(bt / T_);
    float v = (t == 0) ? cls[f] : tok[((long)b * NP + (t - 1)) * E_ + f];
    Hh[i] = v + pos[(long)t * E_ + f];
}

__global__ void gather_cls_kernel(const float* __restrict__ XN, float* __restrict__ CLS) {
    int i = blockIdx.x * 256 + threadIdx.x;
    if (i >= B_ * E_) return;
    int b = i / E_, f = i % E_;
    CLS[i] = XN[(long)b * T_ * E_ + f];
}

// ---------------- host orchestration ----------------
extern "C" void kernel_launch(void* const* d_in, const int* in_sizes, int n_in,
                              void* d_out, int out_size, void* d_ws, size_t ws_size,
                              hipStream_t stream)
{
    (void)in_sizes; (void)n_in; (void)out_size; (void)ws_size;
    const float* x      = (const float*)d_in[0];
    const float* conv_w = (const float*)d_in[1];
    const float* conv_b = (const float*)d_in[2];
    const float* cls    = (const float*)d_in[3];
    const float* pos    = (const float*)d_in[4];
    const float* Wq     = (const float*)d_in[5];
    const float* Wk     = (const float*)d_in[6];
    const float* Wv     = (const float*)d_in[7];
    const float* proj_w = (const float*)d_in[8];
    const float* proj_b = (const float*)d_in[9];
    const float* ln1_g  = (const float*)d_in[10];
    const float* ln1_b  = (const float*)d_in[11];
    const float* ln2_g  = (const float*)d_in[12];
    const float* ln2_b  = (const float*)d_in[13];
    const float* ff1_w  = (const float*)d_in[14];
    const float* ff1_b  = (const float*)d_in[15];
    const float* ff2_w  = (const float*)d_in[16];
    const float* ff2_b  = (const float*)d_in[17];
    const float* lnf_g  = (const float*)d_in[18];
    const float* lnf_b  = (const float*)d_in[19];
    const float* head_w = (const float*)d_in[20];
    const float* head_b = (const float*)d_in[21];
    float* out = (float*)d_out;

    // f32 workspace (~388 MB)
    float* ws = (float*)d_ws;
    size_t o = 0;
    float* Hbuf = ws + o; o += (size_t)BT * E_;
    float* XN   = ws + o; o += (size_t)BT * E_;
    float* AO   = ws + o; o += (size_t)BT * E_;
    float* Qb   = ws + o; o += (size_t)H_ * BT * HS_;
    float* Kb   = ws + o; o += (size_t)H_ * BT * HS_;
    float* Vb   = ws + o; o += (size_t)H_ * BT * HS_;
    float* FFb  = ws + o; o += (size_t)BT * FF_;
    float* CLS  = ws + o; o += (size_t)B_ * E_;

    // bf16 swizzled-weight workspace (~173 MB)
    unsigned short* wsu = (unsigned short*)(ws + o);
    size_t u = 0;
    unsigned short* convS = wsu + u; u += (size_t)E_ * E_;
    unsigned short* WqS   = wsu + u; u += (size_t)L_ * H_ * E_ * HS_;
    unsigned short* WkS   = wsu + u; u += (size_t)L_ * H_ * E_ * HS_;
    unsigned short* WvS   = wsu + u; u += (size_t)L_ * H_ * E_ * HS_;
    unsigned short* projS = wsu + u; u += (size_t)L_ * E_ * E_;
    unsigned short* ff1S  = wsu + u; u += (size_t)L_ * E_ * FF_;
    unsigned short* ff2S  = wsu + u; u += (size_t)L_ * FF_ * E_;
    unsigned short* headS = wsu + u; u += (size_t)E_ * NCP;

    dim3 blk(256);
    // ---- one-time weight pre-swizzles (batched over grid.y) ----
    {
        long jobs = (long)E_ * (E_ >> 3);           // conv (transposed)
        swizzle_b_kernel<<<dim3((unsigned)((jobs + 255) / 256), 1, 1), blk, 0, stream>>>(
            conv_w, convS, E_, E_, E_, E_, 1, 0, 0);
        jobs = (long)64 * (E_ >> 3);                // qkv per (l,h): [768 x 64]
        dim3 gq((unsigned)((jobs + 255) / 256), L_ * H_, 1);
        swizzle_b_kernel<<<gq, blk, 0, stream>>>(Wq, WqS, E_, HS_, 64, HS_, 0,
                                                 (long)E_ * HS_, (long)E_ * HS_);
        swizzle_b_kernel<<<gq, blk, 0, stream>>>(Wk, WkS, E_, HS_, 64, HS_, 0,
                                                 (long)E_ * HS_, (long)E_ * HS_);
        swizzle_b_kernel<<<gq, blk, 0, stream>>>(Wv, WvS, E_, HS_, 64, HS_, 0,
                                                 (long)E_ * HS_, (long)E_ * HS_);
        jobs = (long)E_ * (E_ >> 3);                // proj per layer
        swizzle_b_kernel<<<dim3((unsigned)((jobs + 255) / 256), L_, 1), blk, 0, stream>>>(
            proj_w, projS, E_, E_, E_, E_, 0, (long)E_ * E_, (long)E_ * E_);
        jobs = (long)FF_ * (E_ >> 3);               // ff1 per layer
        swizzle_b_kernel<<<dim3((unsigned)((jobs + 255) / 256), L_, 1), blk, 0, stream>>>(
            ff1_w, ff1S, E_, FF_, FF_, FF_, 0, (long)E_ * FF_, (long)E_ * FF_);
        jobs = (long)E_ * (FF_ >> 3);               // ff2 per layer
        swizzle_b_kernel<<<dim3((unsigned)((jobs + 255) / 256), L_, 1), blk, 0, stream>>>(
            ff2_w, ff2S, FF_, E_, E_, E_, 0, (long)FF_ * E_, (long)FF_ * E_);
        jobs = (long)NCP * (E_ >> 3);               // head (N padded to 1024)
        swizzle_b_kernel<<<dim3((unsigned)((jobs + 255) / 256), 1, 1), blk, 0, stream>>>(
            head_w, headS, E_, NC_, NCP, NC_, 0, 0, 0);
    }

    long qkvW = (long)E_ * HS_;       // swizzled per-head stride (ushort elems)
    long qkvC = (long)BT * HS_;
    unsigned gy  = (unsigned)((BT  + GBM - 1) / GBM);   // 99
    unsigned gyp = (unsigned)((BNP + GBM - 1) / GBM);   // 98

    patchify_kernel<<<(unsigned)(((long)BNP * E_ + 255) / 256), blk, 0, stream>>>(x, XN);
    gemm_kernel<<<dim3(E_ / GBN, gyp, 1), blk, 0, stream>>>(
        XN, convS, AO, conv_b, nullptr, BNP, E_, E_, E_, E_, 0, 0, 0, 0);
    build_h0_kernel<<<(unsigned)(((long)BT * E_ + 255) / 256), blk, 0, stream>>>(
        AO, cls, pos, Hbuf);

    for (int l = 0; l < L_; ++l) {
        layernorm_kernel<<<(BT + 7) / 8, blk, 0, stream>>>(
            Hbuf, ln1_g + l * E_, ln1_b + l * E_, XN, BT);

        long wl = (long)l * H_ * E_ * HS_;
        gemm_kernel<<<dim3(1, gy, H_), blk, 0, stream>>>(
            XN, WqS + wl, Qb, nullptr, nullptr,
            BT, HS_, E_, E_, HS_, 0, qkvW, qkvC, 0);
        gemm_kernel<<<dim3(1, gy, H_), blk, 0, stream>>>(
            XN, WkS + wl, Kb, nullptr, nullptr,
            BT, HS_, E_, E_, HS_, 0, qkvW, qkvC, 0);
        gemm_kernel<<<dim3(1, gy, H_), blk, 0, stream>>>(
            XN, WvS + wl, Vb, nullptr, nullptr,
            BT, HS_, E_, E_, HS_, 0, qkvW, qkvC, 0);

        attention_kernel<<<B_ * H_, blk, 0, stream>>>(Qb, Kb, Vb, AO);

        gemm_kernel<<<dim3(E_ / GBN, gy, 1), blk, 0, stream>>>(
            AO, projS + (long)l * E_ * E_, Hbuf, proj_b + l * E_, Hbuf,
            BT, E_, E_, E_, E_, 0, 0, 0, 0);

        layernorm_kernel<<<(BT + 7) / 8, blk, 0, stream>>>(
            Hbuf, ln2_g + l * E_, ln2_b + l * E_, XN, BT);

        gemm_kernel<<<dim3(FF_ / GBN, gy, 1), blk, 0, stream>>>(
            XN, ff1S + (long)l * E_ * FF_, FFb, ff1_b + (long)l * FF_, nullptr,
            BT, FF_, E_, E_, FF_, 0, 0, 0, /*gelu=*/1);

        gemm_kernel<<<dim3(E_ / GBN, gy, 1), blk, 0, stream>>>(
            FFb, ff2S + (long)l * FF_ * E_, Hbuf, ff2_b + l * E_, Hbuf,
            BT, E_, FF_, FF_, E_, 0, 0, 0, 0);
    }

    layernorm_kernel<<<(BT + 7) / 8, blk, 0, stream>>>(Hbuf, lnf_g, lnf_b, XN, BT);
    gather_cls_kernel<<<(B_ * E_ + 255) / 256, blk, 0, stream>>>(XN, CLS);
    gemm_kernel<<<dim3(NCP / GBN, 1, 1), blk, 0, stream>>>(
        CLS, headS, out, head_b, nullptr, B_, NC_, E_, E_, NC_, 0, 0, 0, 0);
}
